// SelfAttentionLayer_47433618817588
// MI455X (gfx1250) — matrix operand order repaired and checked by reading, not compiled, not run on hardware
//
#include <hip/hip_runtime.h>
#include <hip/hip_bf16.h>

// ---------------------------------------------------------------------------
// Self-attention for MI455X (gfx1250), wave32, bf16 WMMA (16x16x32).
//   B=8, S=2048, E=1024.  d_out = [output f32 (8*2048*1024)] ++ [attn f32 (8*2048*2048)]
// Memory-bound (~435MB @ 23.3TB/s => ~19us floor); WMMA compute negligible.
// Round 2: proj kernel restaged (LDS bf16 X-tile, full-N sweep, B-frag reuse x2)
// to cut L2/HBM A-traffic 20x; Q-tile staged via GLOBAL_LOAD_ASYNC_TO_LDS_B128
// (ASYNCcnt) and K/V prefetched one tile ahead (global_prefetch_b8).
// ---------------------------------------------------------------------------

typedef __bf16 bf16;
typedef __attribute__((ext_vector_type(16))) __bf16 v16bf;
typedef __attribute__((ext_vector_type(8)))  __bf16 v8bf;
typedef __attribute__((ext_vector_type(4)))  __bf16 v4bf;
typedef __attribute__((ext_vector_type(8)))  float  v8f;
typedef __attribute__((ext_vector_type(4)))  float  v4f;

#define BATCH 8
#define SEQ   2048
#define EDIM  1024

__device__ __forceinline__ v16bf bf_concat(v8bf lo, v8bf hi) {
  v16bf r;
#pragma unroll
  for (int i = 0; i < 8; ++i) { r[i] = lo[i]; r[8 + i] = hi[i]; }
  return r;
}

__device__ __forceinline__ v16bf bf_from_f32(v4f f0, v4f f1, v4f f2, v4f f3) {
  v16bf r;
#pragma unroll
  for (int i = 0; i < 4; ++i) {
    r[i]      = (bf16)f0[i];
    r[4 + i]  = (bf16)f1[i];
    r[8 + i]  = (bf16)f2[i];
    r[12 + i] = (bf16)f3[i];
  }
  return r;
}

// ---------------------------------------------------------------------------
// Kernel 0: transpose weights to bf16  Wt[mat][n][e] = (bf16) W[mat][e][n]
// so that WMMA B fragments (lane = N column, two runs of 8 contiguous K) are contiguous.
// ---------------------------------------------------------------------------
__global__ void prep_kernel(const float* __restrict__ Wq,
                            const float* __restrict__ Wk,
                            const float* __restrict__ Wv,
                            bf16* __restrict__ Wt) {
  int mat = blockIdx.y;
  int idx = blockIdx.x * 256 + threadIdx.x;         // 0 .. 1M-1
  const float* W = (mat == 0) ? Wq : (mat == 1) ? Wk : Wv;
  int e = idx >> 10;        // coalesced read along n
  int n = idx & 1023;
  Wt[((size_t)mat << 20) + ((size_t)n << 10) + e] = (bf16)W[((size_t)e << 10) + n];
}

// ---------------------------------------------------------------------------
// Kernel 1: Q/K/V projections.  grid = (M/32, 3), block = 128 (4 waves).
// One workgroup: 32 M-rows x all 1024 N.  X tile staged once in LDS as bf16
// (read 1x from HBM); each wave sweeps 16 N-tiles; each Wt B-fragment is
// reused by two WMMAs (two 16-row accumulators).
// mat 0/1 -> Q/K bf16 row-major [B*S][E];  mat 2 -> Vt bf16 [B][E][S]
// ---------------------------------------------------------------------------
__global__ void __launch_bounds__(128) proj_kernel(
    const float* __restrict__ Xq, const float* __restrict__ Xk,
    const float* __restrict__ Xv,
    const float* __restrict__ bq, const float* __restrict__ bk,
    const float* __restrict__ bv,
    const bf16* __restrict__ Wt,
    bf16* __restrict__ Qb, bf16* __restrict__ Kb, bf16* __restrict__ Vt) {
  __shared__ bf16 xt[32 * EDIM];                       // 64 KB
  const int mat = blockIdx.y;
  const float* X    = (mat == 0) ? Xq : (mat == 1) ? Xk : Xv;
  const float* bias = (mat == 0) ? bq : (mat == 1) ? bk : bv;
  const bf16*  W    = Wt + ((size_t)mat << 20);

  const int tid  = threadIdx.x;
  const int wave = tid >> 5;
  const int lane = tid & 31;
  const int l16  = lane & 15;
  const int cs   = lane >> 4;                          // half-wave select
  const int m0   = blockIdx.x * 32;

  // ---- stage 32 rows of X, f32 -> bf16, one pass ---------------------------
  {
    const v4f* xsrc = (const v4f*)(X + (size_t)m0 * EDIM);  // rows contiguous
    for (int i = tid; i < 8192; i += 128) {                 // 8192 float4
      v4f f = xsrc[i];
      v4bf c;
#pragma unroll
      for (int j = 0; j < 4; ++j) c[j] = (bf16)f[j];
      ((v4bf*)xt)[i] = c;
    }
  }
  __syncthreads();

  const bf16* arow0 = xt + l16 * EDIM;                 // A rows 0..15
  const bf16* arow1 = xt + (16 + l16) * EDIM;          // A rows 16..31

  for (int nt = wave; nt < 64; nt += 4) {
    const int n0 = nt * 16;
    const bf16* brow = W + (size_t)(n0 + l16) * EDIM;
    v8f acc0 = {}, acc1 = {};
#pragma unroll 2
    for (int kk = 0; kk < 32; ++kk) {
      const int e0 = kk * 32;
      v8bf b0 = *(const v8bf*)(brow + e0 + cs * 8);
      v8bf b1 = *(const v8bf*)(brow + e0 + 16 + cs * 8);
      v16bf bb = bf_concat(b0, b1);
      v16bf a0 = bf_concat(*(const v8bf*)(arow0 + e0 + cs * 8),
                           *(const v8bf*)(arow0 + e0 + 16 + cs * 8));
      v16bf a1 = bf_concat(*(const v8bf*)(arow1 + e0 + cs * 8),
                           *(const v8bf*)(arow1 + e0 + 16 + cs * 8));
      acc0 = __builtin_amdgcn_wmma_f32_16x16x32_bf16(false, a0, false, bb,
                                                     (short)0, acc0, false, false);
      acc1 = __builtin_amdgcn_wmma_f32_16x16x32_bf16(false, a1, false, bb,
                                                     (short)0, acc1, false, false);
    }
    const float bn = bias[n0 + l16];
    if (mat < 2) {
      bf16* Y = (mat == 0) ? Qb : Kb;
#pragma unroll
      for (int v = 0; v < 8; ++v) {
        int m = m0 + v + 8 * cs;                       // C layout: m = vgpr + 8*cs
        Y[(size_t)m * EDIM + n0 + l16]        = (bf16)(acc0[v] + bn);
        Y[(size_t)(m + 16) * EDIM + n0 + l16] = (bf16)(acc1[v] + bn);
      }
    } else {
      // Vt[b][e][s]; lane's 8 C values are 8 consecutive s -> one 16B store
      const int bb2 = m0 >> 11;                        // m0 / SEQ (tile stays in batch)
      const int s0  = (m0 & (SEQ - 1)) + 8 * cs;
      v8bf pk0, pk1;
#pragma unroll
      for (int v = 0; v < 8; ++v) {
        pk0[v] = (bf16)(acc0[v] + bn);
        pk1[v] = (bf16)(acc1[v] + bn);
      }
      bf16* vdst = Vt + ((size_t)bb2 * EDIM + n0 + l16) * SEQ;
      *(v8bf*)(vdst + s0)      = pk0;
      *(v8bf*)(vdst + s0 + 16) = pk1;
    }
  }
}

// ---------------------------------------------------------------------------
// Kernel 2: fused scores + mask + softmax + attn@V.
// grid = (S/16, B), block = 128 (4 waves), one 16-query tile per workgroup.
// Dynamic LDS: qtile bf16 16x1024 (32KB) | scores f32 16x2048 (128KB) |
//              red[128] | rstat[16]   => 164416 bytes.
// ---------------------------------------------------------------------------
__global__ void __launch_bounds__(128) attn_kernel(
    const bf16* __restrict__ Qb, const bf16* __restrict__ Kb,
    const bf16* __restrict__ Vt, const unsigned char* __restrict__ mask,
    float* __restrict__ out, float* __restrict__ attn) {
  extern __shared__ char smem[];
  bf16*  qt    = (bf16*)smem;                         // 16 x 1024 bf16
  float* sc    = (float*)(smem + 32768);              // 16 x 2048 f32
  float* red   = (float*)(smem + 32768 + 131072);     // 128 f32
  float* rstat = red + 128;                           // 16 f32 (rowmax, then 1/sum)

  const int tid  = threadIdx.x;
  const int wave = tid >> 5;
  const int lane = tid & 31;
  const int l16  = lane & 15;
  const int cs   = lane >> 4;

  const int    qb0     = blockIdx.x * 16;             // query tile within batch
  const int    b       = blockIdx.y;
  const size_t rowbase = (size_t)b * SEQ;

  // ---- stage Q tile via async global->LDS (ASYNCcnt), 32KB contiguous ----
  {
    const uint4* src = (const uint4*)(Qb + (rowbase + qb0) * EDIM);
    const unsigned ldsbase = (unsigned)(unsigned long long)(void*)qt;
    for (int i = tid; i < 2048; i += 128) {
      unsigned laddr = ldsbase + (unsigned)i * 16u;
      unsigned long long gaddr = (unsigned long long)(src + i);
      asm volatile("global_load_async_to_lds_b128 %0, %1, off"
                   :: "v"(laddr), "v"(gaddr) : "memory");
    }
    asm volatile("s_wait_asynccnt 0x0" ::: "memory");
  }
  __syncthreads();

  // ---- phase 1: scores = Q @ K^T, mask, scale, -> LDS --------------------
  const bf16* Kbase = Kb + rowbase * EDIM;
  for (int nt = wave; nt < 128; nt += 4) {
    const int k0 = nt * 16;
    const bf16* krow = Kbase + (size_t)(k0 + l16) * EDIM;
    const bf16* qrow = qt + l16 * EDIM;
    if (nt + 4 < 128) {                               // prefetch next K tile row
      __builtin_prefetch(Kbase + (size_t)(k0 + 64 + l16) * EDIM, 0, 1);
    }
    v8f acc = {};
#pragma unroll 4
    for (int kk = 0; kk < 32; ++kk) {
      const int e0 = kk * 32;
      v8bf a0 = *(const v8bf*)(qrow + e0 + cs * 8);
      v8bf a1 = *(const v8bf*)(qrow + e0 + 16 + cs * 8);
      v8bf b0 = *(const v8bf*)(krow + e0 + cs * 8);
      v8bf b1 = *(const v8bf*)(krow + e0 + 16 + cs * 8);
      acc = __builtin_amdgcn_wmma_f32_16x16x32_bf16(false, bf_concat(a0, a1),
                                                    false, bf_concat(b0, b1),
                                                    (short)0, acc, false, false);
    }
    const unsigned char* mrow = mask + ((size_t)b * SEQ + qb0) * SEQ + k0;
#pragma unroll
    for (int v = 0; v < 8; ++v) {
      int m = v + 8 * cs;
      unsigned char mk = mrow[(size_t)m * SEQ + l16];
      // reference: where(mask, s, -1e32) / sqrt(1024)
      sc[m * SEQ + k0 + l16] = mk ? acc[v] * 0.03125f : -3.125e30f;
    }
  }
  __syncthreads();

  // ---- phase 2: softmax over 2048 (8 threads per row) --------------------
  const int row = tid >> 3, seg = tid & 7;
  float* srow = sc + row * SEQ + seg * 256;
  float lmax = -3.4e38f;
  for (int i = 0; i < 256; ++i) lmax = fmaxf(lmax, srow[i]);
  red[tid] = lmax;
  __syncthreads();
  if (seg == 0) {
    float m = red[tid];
    for (int j = 1; j < 8; ++j) m = fmaxf(m, red[tid + j]);
    rstat[row] = m;
  }
  __syncthreads();
  const float rm = rstat[row];
  float lsum = 0.f;
  for (int i = 0; i < 256; ++i) {
    float e = __expf(srow[i] - rm);
    srow[i] = e;                      // keep unnormalized e in LDS
    lsum += e;
  }
  __syncthreads();                    // rstat[row] consumed by all before overwrite
  red[tid] = lsum;
  __syncthreads();
  if (seg == 0) {
    float s = 0.f;
    for (int j = 0; j < 8; ++j) s += red[tid + j];
    rstat[row] = 1.0f / s;            // fold normalization into outputs
  }
  __syncthreads();

  // ---- write normalized attention weights, coalesced ---------------------
  float* adst = attn + ((size_t)b * SEQ + qb0) * SEQ;  // 16 rows contiguous
  for (int i = tid; i < 8192; i += 128) {              // 8192 float4 = 16*2048
    const float inv = rstat[i >> 9];
    v4f e = ((const v4f*)sc)[i];
    v4f o;
#pragma unroll
    for (int j = 0; j < 4; ++j) o[j] = e[j] * inv;
    ((v4f*)adst)[i] = o;
  }

  // ---- phase 3: out = (e @ V) * (1/rowsum) -------------------------------
  const bf16* Vbase = Vt + (size_t)b * EDIM * SEQ;     // Vt[b][e][s]
  float* obase = out + (rowbase + qb0) * EDIM;
  for (int et = wave; et < 64; et += 4) {
    const int n0 = et * 16;
    const bf16*  vrow = Vbase + (size_t)(n0 + l16) * SEQ;
    const float* srA  = sc + l16 * SEQ;
    if (et + 4 < 64) {                                 // prefetch next V tile row
      __builtin_prefetch(Vbase + (size_t)(n0 + 64 + l16) * SEQ, 0, 1);
    }
    v8f acc = {};
#pragma unroll 4
    for (int kc = 0; kc < 64; ++kc) {
      const int k0 = kc * 32;
      v4f f0 = *(const v4f*)(srA + k0 + cs * 8);
      v4f f1 = *(const v4f*)(srA + k0 + cs * 8 + 4);
      v4f f2 = *(const v4f*)(srA + k0 + 16 + cs * 8);
      v4f f3 = *(const v4f*)(srA + k0 + 16 + cs * 8 + 4);
      v16bf a = bf_from_f32(f0, f1, f2, f3);
      v8bf b0 = *(const v8bf*)(vrow + k0 + cs * 8);
      v8bf b1 = *(const v8bf*)(vrow + k0 + 16 + cs * 8);
      acc = __builtin_amdgcn_wmma_f32_16x16x32_bf16(false, a, false,
                                                    bf_concat(b0, b1),
                                                    (short)0, acc, false, false);
    }
#pragma unroll
    for (int v = 0; v < 8; ++v) {
      int m = v + 8 * cs;
      obase[(size_t)m * EDIM + n0 + l16] = acc[v] * rstat[m];
    }
  }
}

// ---------------------------------------------------------------------------
extern "C" void kernel_launch(void* const* d_in, const int* in_sizes, int n_in,
                              void* d_out, int out_size, void* d_ws, size_t ws_size,
                              hipStream_t stream) {
  const float* qx = (const float*)d_in[0];
  const float* kx = (const float*)d_in[1];
  const float* vx = (const float*)d_in[2];
  const unsigned char* mask = (const unsigned char*)d_in[3];  // jnp.bool_ = 1 byte
  const float* Wq = (const float*)d_in[4];
  const float* bq = (const float*)d_in[5];
  const float* Wk = (const float*)d_in[6];
  const float* bk = (const float*)d_in[7];
  const float* Wv = (const float*)d_in[8];
  const float* bv = (const float*)d_in[9];

  float* out  = (float*)d_out;                               // [B*S*E]
  float* attn = out + (size_t)BATCH * SEQ * EDIM;            // [B*S*S]

  // workspace layout (bf16): Wt[3][1024][1024] | Q[B*S][E] | K[B*S][E] | Vt[B][E][S]
  char* ws = (char*)d_ws;
  bf16* Wt = (bf16*)ws;
  bf16* Qb = (bf16*)(ws + (size_t)3 * EDIM * EDIM * 2);
  bf16* Kb = Qb + (size_t)BATCH * SEQ * EDIM;
  bf16* Vt = Kb + (size_t)BATCH * SEQ * EDIM;

  prep_kernel<<<dim3((EDIM * EDIM) / 256, 3), 256, 0, stream>>>(Wq, Wk, Wv, Wt);

  proj_kernel<<<dim3((BATCH * SEQ) / 32, 3), 128, 0, stream>>>(
      qx, kx, vx, bq, bk, bv, Wt, Qb, Kb, Vt);

  const size_t lds_bytes = 32768 + 131072 + (128 + 16) * sizeof(float);  // 164416
  attn_kernel<<<dim3(SEQ / 16, BATCH), 128, lds_bytes, stream>>>(
      Qb, Kb, Vt, mask, out, attn);
}
